// MultiHeadAttention_1211180778266
// MI455X (gfx1250) — matrix operand order repaired and checked by reading, not compiled, not run on hardware
//
#include <hip/hip_runtime.h>

// ---------------------------------------------------------------------------
// MultiHeadAttention on MI455X (gfx1250, wave32, WMMA bf16 16x16x32, f32 acc)
//   S=2048 seq, E=1024 embed, H=16 heads, D=64 head dim.
// ~30 GFLOP vs ~30MB unique data -> compute-bound: all matmuls via
// v_wmma_f32_16x16x32_bf16. The 192MB L2 holds the full bf16 working set
// (~22MB) -> direct global fragment loads (no LDS tiling of GEMM operands).
// LDS only for the flash-attention P-tile C-layout -> A-layout relayout
// (per-wave, barrier-free; same-wave DS ops are in-order).
// Proj K-loop is software-pipelined (ping-pong fragment buffers) so
// global_load_b128 clauses overlap WMMA issue instead of s_wait_loadcnt 0.
// ---------------------------------------------------------------------------

typedef __attribute__((ext_vector_type(16))) __bf16 v16bf;
typedef __attribute__((ext_vector_type(8)))  __bf16 v8bf;
typedef __attribute__((ext_vector_type(8)))  float  v8f;

#define SEQ  2048
#define EMB  1024
#define NH   16
#define HD   64

__device__ __forceinline__ unsigned short f2bf(float f) {
    union { float f; unsigned u; } c; c.f = f;
    unsigned u = c.u + 0x7FFFu + ((c.u >> 16) & 1u);   // round-to-nearest-even
    return (unsigned short)(u >> 16);
}

__device__ __forceinline__ unsigned pack2bf(float lo, float hi) {
    return (unsigned)f2bf(lo) | ((unsigned)f2bf(hi) << 16);
}

__device__ __forceinline__ v8f vzero8() {
    v8f z = {0.f,0.f,0.f,0.f,0.f,0.f,0.f,0.f};
    return z;
}

__device__ __forceinline__ v16bf combine16(v8bf lo, v8bf hi) {
    return __builtin_shufflevector(lo, hi, 0,1,2,3,4,5,6,7,8,9,10,11,12,13,14,15);
}

// ---------------------------------------------------------------------------
// Kernel 1: f32 -> bf16 conversion (2 elements/thread, packed b32 stores)
// ---------------------------------------------------------------------------
__global__ void mha_cvt_kernel(const float* __restrict__ x,
                               const float* __restrict__ wq,
                               const float* __restrict__ wk,
                               const float* __restrict__ wv,
                               unsigned* __restrict__ xb,
                               unsigned* __restrict__ wqb,
                               unsigned* __restrict__ wkb,
                               unsigned* __restrict__ wvb) {
    const unsigned NX = (SEQ * EMB) / 2;   // in pairs
    const unsigned NW = (EMB * EMB) / 2;
    const unsigned g = blockIdx.x * blockDim.x + threadIdx.x;
    const float* src; unsigned* dst; unsigned off;
    if (g < NX)               { src = x;  dst = xb;  off = g; }
    else if (g < NX + NW)     { src = wq; dst = wqb; off = g - NX; }
    else if (g < NX + 2*NW)   { src = wk; dst = wkb; off = g - NX - NW; }
    else                      { src = wv; dst = wvb; off = g - NX - 2*NW; }
    const float2 f = reinterpret_cast<const float2*>(src)[off];
    dst[off] = pack2bf(f.x, f.y);
}

// ---------------------------------------------------------------------------
// Projection fragment helpers (ping-pong software pipeline)
//   A-fragment (16-bit 16x32): lane=M, elem j <-> K = half*16 + j (contiguous)
//   B-fragment (32x16): lane=N, elem j <-> K = half*16 + j (contiguous)
// ---------------------------------------------------------------------------
__device__ __forceinline__ void proj_load_frags(const unsigned short* __restrict__ ap,
                                                const unsigned short* __restrict__ bp,
                                                int k0, int half,
                                                v16bf a[4], v16bf b[2]) {
    #pragma unroll
    for (int i = 0; i < 4; i++) {
        const unsigned short* p = ap + i * 16 * EMB + k0;
        v8bf lo = *reinterpret_cast<const v8bf*>(p + half * 16);
        v8bf hi = *reinterpret_cast<const v8bf*>(p + half * 16 + 8);
        a[i] = combine16(lo, hi);
    }
    #pragma unroll
    for (int j = 0; j < 2; j++)
        b[j] = *reinterpret_cast<const v16bf*>(bp + j * 16 * EMB + k0);
}

__device__ __forceinline__ void proj_wmma8(const v16bf a[4], const v16bf b[2],
                                           v8f acc[4][2]) {
    #pragma unroll
    for (int i = 0; i < 4; i++)
        #pragma unroll
        for (int j = 0; j < 2; j++)
            acc[i][j] = __builtin_amdgcn_wmma_f32_16x16x32_bf16(
                false, a[i], false, b[j], (short)0, acc[i][j], false, false);
}

// ---------------------------------------------------------------------------
// Kernel 2: projections  C[s,o] = sum_e xb[s,e] * Wb[o,e]
//   grid (EMB/128, SEQ/128, 3), 256 threads = 8 waves.
//   Wave computes 64(M) x 32(N): 4x2 accum tiles, K-step 32, 8 WMMAs/step.
//   mat 0 -> Qh[h][s][d], mat 1 -> Kh[h][s][d], mat 2 -> Vt[o][s] (= [h][d][s])
// ---------------------------------------------------------------------------
__global__ __launch_bounds__(256, 1)
void mha_proj_kernel(const unsigned short* __restrict__ xb,
                     const unsigned short* __restrict__ wqb,
                     const unsigned short* __restrict__ wkb,
                     const unsigned short* __restrict__ wvb,
                     unsigned short* __restrict__ qh,
                     unsigned short* __restrict__ kh,
                     unsigned short* __restrict__ vt) {
    const int lane = threadIdx.x & 31;
    const int w    = threadIdx.x >> 5;
    const int ln   = lane & 15;
    const int half = lane >> 4;
    const int n0 = blockIdx.x * 128 + (w >> 1) * 32;   // N base for this wave
    const int m0 = blockIdx.y * 128 + (w & 1) * 64;    // M base for this wave
    const int mat = blockIdx.z;
    const unsigned short* Wb = (mat == 0) ? wqb : (mat == 1) ? wkb : wvb;

    const unsigned short* ap = xb + (m0 + ln) * EMB;              // per-lane A base
    const unsigned short* bp = Wb + (n0 + ln) * EMB + half * 16;  // per-lane B base

    v8f acc[4][2];
    #pragma unroll
    for (int i = 0; i < 4; i++)
        #pragma unroll
        for (int j = 0; j < 2; j++) acc[i][j] = vzero8();

    // software-pipelined K loop: ping-pong fragment buffers, unroll-by-2
    v16bf aA[4], bA[2], aB[4], bB[2];
    proj_load_frags(ap, bp, 0, half, aA, bA);
    for (int k0 = 0; k0 < EMB; k0 += 64) {
        proj_load_frags(ap, bp, k0 + 32, half, aB, bB);
        proj_wmma8(aA, bA, acc);
        const int knext = (k0 + 64 < EMB) ? (k0 + 64) : (EMB - 32); // branchless tail
        proj_load_frags(ap, bp, knext, half, aA, bA);
        proj_wmma8(aB, bB, acc);
    }

    unsigned short* qk = (mat == 0) ? qh : kh;
    if (mat == 2) {
        // V transposed [h*64+d][s]: consecutive r -> consecutive s -> pack b32
        #pragma unroll
        for (int i = 0; i < 4; i++)
            #pragma unroll
            for (int j = 0; j < 2; j++) {
                const int o = n0 + j*16 + ln;
                const int s = m0 + i*16 + 8*half;      // row base (r=0)
                unsigned* dst = reinterpret_cast<unsigned*>(vt + o * SEQ + s);
                #pragma unroll
                for (int r = 0; r < 8; r += 2)
                    dst[r >> 1] = pack2bf(acc[i][j][r], acc[i][j][r + 1]);
            }
    } else {
        #pragma unroll
        for (int i = 0; i < 4; i++)
            #pragma unroll
            for (int j = 0; j < 2; j++)
                #pragma unroll
                for (int r = 0; r < 8; r++) {
                    const int s = m0 + i*16 + r + 8*half;   // C-layout row
                    const int o = n0 + j*16 + ln;           // C-layout col
                    qk[(o >> 6) * (SEQ*HD) + s * HD + (o & 63)] = f2bf(acc[i][j][r]);
                }
    }
}

// ---------------------------------------------------------------------------
// Kernel 3: flash attention per (head, 128-query block); 8 waves, wave owns
// 16 query rows. Key step = 32: 4 WMMAs (scores) + online softmax + 4 WMMAs (PV).
// V fragments are loaded BEFORE the softmax VALU block so exp/shuffle work
// hides their latency.
// ---------------------------------------------------------------------------
__global__ __launch_bounds__(256, 1)
void mha_attn_kernel(const unsigned short* __restrict__ qh,
                     const unsigned short* __restrict__ kh,
                     const unsigned short* __restrict__ vt,
                     float* __restrict__ out) {
    __shared__ __align__(16) unsigned short Plds[8 * 512];   // 16x32 bf16 per wave

    const int lane = threadIdx.x & 31;
    const int w    = threadIdx.x >> 5;
    const int ln   = lane & 15;
    const int half = lane >> 4;
    const int h     = blockIdx.x;
    const int qbase = blockIdx.y * 128 + w * 16;

    const unsigned short* Qp = qh + h * (SEQ * HD);
    const unsigned short* Kp = kh + h * (SEQ * HD) + (ln * HD) + half * 16;
    const unsigned short* Vp = vt + h * (HD * SEQ) + (ln * SEQ) + half * 16;
    unsigned short* Pl = &Plds[w * 512];

    // Q fragments (resident): dstep 0 -> d 0..31, dstep 1 -> d 32..63
    v16bf qf[2];
    #pragma unroll
    for (int ds = 0; ds < 2; ds++) {
        const unsigned short* p = Qp + (qbase + ln) * HD + ds * 32;
        v8bf lo = *reinterpret_cast<const v8bf*>(p + half * 16);
        v8bf hi = *reinterpret_cast<const v8bf*>(p + half * 16 + 8);
        qf[ds] = combine16(lo, hi);
    }

    v8f o[4];
    #pragma unroll
    for (int dt = 0; dt < 4; dt++) o[dt] = vzero8();
    float mstat[8], lstat[8];
    #pragma unroll
    for (int r = 0; r < 8; r++) { mstat[r] = -1.0e30f; lstat[r] = 0.0f; }

    const float scale = 0.125f;           // 1/sqrt(64)
    const int kend = qbase + 15;          // causal bound for this wave

    for (int k0 = 0; k0 <= kend; k0 += 32) {
        // ---- K fragments for 32 keys, then 4 score WMMAs ----
        v16bf kf[2][2];
        #pragma unroll
        for (int t = 0; t < 2; t++)
            #pragma unroll
            for (int ds = 0; ds < 2; ds++)
                kf[t][ds] = *reinterpret_cast<const v16bf*>(
                    Kp + (k0 + t*16) * HD + ds * 32);
        v8f sc[2];
        sc[0] = vzero8(); sc[1] = vzero8();
        #pragma unroll
        for (int t = 0; t < 2; t++)
            #pragma unroll
            for (int ds = 0; ds < 2; ds++)
                sc[t] = __builtin_amdgcn_wmma_f32_16x16x32_bf16(
                    false, qf[ds], false, kf[t][ds], (short)0, sc[t], false, false);

        // ---- issue V loads early: softmax VALU below hides their latency ----
        v16bf vf[4];
        #pragma unroll
        for (int dt = 0; dt < 4; dt++)
            vf[dt] = *reinterpret_cast<const v16bf*>(Vp + (dt*16) * SEQ + k0);

        // ---- scale + causal mask + online softmax update ----
        #pragma unroll
        for (int r = 0; r < 8; r++) {
            const int row = qbase + r + 8 * half;
            float mx = -1.0e30f;
            #pragma unroll
            for (int t = 0; t < 2; t++) {
                const int key = k0 + t*16 + ln;
                float v = sc[t][r] * scale;
                v = (key > row) ? -1.0e30f : v;
                sc[t][r] = v;
                mx = fmaxf(mx, v);
            }
            // row reduction across the 16 lanes of this half (wave32)
            mx = fmaxf(mx, __shfl_xor(mx, 1));
            mx = fmaxf(mx, __shfl_xor(mx, 2));
            mx = fmaxf(mx, __shfl_xor(mx, 4));
            mx = fmaxf(mx, __shfl_xor(mx, 8));
            const float mnew = fmaxf(mstat[r], mx);
            const float corr = __expf(mstat[r] - mnew);
            mstat[r] = mnew;
            float rs = 0.0f;
            #pragma unroll
            for (int t = 0; t < 2; t++) {
                const float p = __expf(sc[t][r] - mnew);
                sc[t][r] = p;
                rs += p;
            }
            rs += __shfl_xor(rs, 1);
            rs += __shfl_xor(rs, 2);
            rs += __shfl_xor(rs, 4);
            rs += __shfl_xor(rs, 8);
            lstat[r] = lstat[r] * corr + rs;
            #pragma unroll
            for (int dt = 0; dt < 4; dt++) o[dt][r] *= corr;
            // spill P (bf16) to per-wave LDS in row-major 16x32
            #pragma unroll
            for (int t = 0; t < 2; t++)
                Pl[(r + 8*half) * 32 + t*16 + ln] = f2bf(sc[t][r]);
        }

        // ---- reload P as A-fragment (same-wave DS ops are in-order) ----
        v8bf plo = *reinterpret_cast<const v8bf*>(&Pl[ln * 32 + half * 16]);
        v8bf phi = *reinterpret_cast<const v8bf*>(&Pl[ln * 32 + half * 16 + 8]);
        v16bf pf = combine16(plo, phi);

        // ---- O += P * V  (V transposed: per-lane contiguous along keys) ----
        #pragma unroll
        for (int dt = 0; dt < 4; dt++)
            o[dt] = __builtin_amdgcn_wmma_f32_16x16x32_bf16(
                false, pf, false, vf[dt], (short)0, o[dt], false, false);
    }

    // ---- normalize and write out[s][h*64+d] (f32) ----
    #pragma unroll
    for (int r = 0; r < 8; r++) {
        const float inv = 1.0f / lstat[r];
        const int row = qbase + r + 8 * half;
        #pragma unroll
        for (int dt = 0; dt < 4; dt++)
            out[row * (NH * HD) + h * HD + dt*16 + ln] = o[dt][r] * inv;
    }
}

// ---------------------------------------------------------------------------
// Host launcher
// ---------------------------------------------------------------------------
extern "C" void kernel_launch(void* const* d_in, const int* in_sizes, int n_in,
                              void* d_out, int out_size, void* d_ws, size_t ws_size,
                              hipStream_t stream) {
    (void)in_sizes; (void)n_in; (void)out_size; (void)ws_size;
    const float* x  = (const float*)d_in[0];
    const float* wq = (const float*)d_in[1];
    const float* wk = (const float*)d_in[2];
    const float* wv = (const float*)d_in[3];
    float* out = (float*)d_out;

    // workspace layout (bf16 stored as u16), ~22 MB total
    char* ws = (char*)d_ws;
    unsigned short* xb  = (unsigned short*)(ws);                         // 4 MB
    unsigned short* wqb = (unsigned short*)(ws + (4u  << 20));           // 2 MB
    unsigned short* wkb = (unsigned short*)(ws + (6u  << 20));           // 2 MB
    unsigned short* wvb = (unsigned short*)(ws + (8u  << 20));           // 2 MB
    unsigned short* qh  = (unsigned short*)(ws + (10u << 20));           // 4 MB
    unsigned short* kh  = (unsigned short*)(ws + (14u << 20));           // 4 MB
    unsigned short* vt  = (unsigned short*)(ws + (18u << 20));           // 4 MB

    const unsigned npair = (SEQ*EMB + 3*EMB*EMB) / 2;   // 2,621,440 pairs
    mha_cvt_kernel<<<(npair + 255) / 256, 256, 0, stream>>>(
        x, wq, wk, wv, (unsigned*)xb, (unsigned*)wqb, (unsigned*)wkb, (unsigned*)wvb);

    mha_proj_kernel<<<dim3(EMB / 128, SEQ / 128, 3), 256, 0, stream>>>(
        xb, wqb, wkb, wvb, qh, kh, vt);

    mha_attn_kernel<<<dim3(NH, SEQ / 128), 256, 0, stream>>>(
        qh, kh, vt, out);
}
